// HierarchicalMultiHeadAttention_30511447671478
// MI455X (gfx1250) — matrix operand order repaired and checked by reading, not compiled, not run on hardware
//
#include <hip/hip_runtime.h>
#include <stdint.h>

// ---------------- problem constants (from reference) ----------------
constexpr int Bc  = 4;
constexpr int LQc = 384;
constexpr int LKc = 384;
constexpr int NLc = 6;
constexpr int Dc  = 1024;
constexpr int Hc  = 16;
constexpr int DHc = 64;

typedef __attribute__((ext_vector_type(16))) __bf16 v16bf;
typedef __attribute__((ext_vector_type(8)))  float  v8f;

union FragU { v16bf v; uint32_t u[8]; };

__device__ __forceinline__ uint32_t f2bf(float f) {
  uint32_t x = __float_as_uint(f);
  return ((x + 0x7FFFu + ((x >> 16) & 1u)) >> 16) & 0xFFFFu;   // RNE f32->bf16
}
__device__ __forceinline__ uint32_t pack2bf(float lo, float hi) {
  return f2bf(lo) | (f2bf(hi) << 16);
}

// =====================================================================
// Generic bf16 WMMA GEMM:  C[M,N] = rowscale(row) * (A[M,K] @ W[K,N])
//   rowscale(row) = cscale * rmask[(row/MDIV) % MMOD]   (compile-time MDIV/MMOD)
// Requires M%128==0, N%64==0, K%64==0 (true for all uses here).
// WG = 256 threads = 8 waves in a 4(M) x 2(N) grid; 32x32 per wave.
// K-step = 64 -> 8 WMMAs per wave between barriers.
// =====================================================================
template <int MDIV, int MMOD>
__global__ __launch_bounds__(256)
void gemm_bf16(const float* __restrict__ A, const float* __restrict__ W,
               float* __restrict__ C, int K, int N,
               float cscale, const int* __restrict__ rmask)
{
  __shared__ uint32_t Alds[128][33];  // 128 rows x 64 k (bf16 pairs), padded
  __shared__ uint32_t Bt[64][33];     // 64 n x 64 k (bf16 pairs), padded

  const int tid  = threadIdx.x;
  const int wave = tid >> 5;
  const int lane = tid & 31;
  const int g    = lane >> 4;
  const int r    = lane & 15;
  const int wm   = wave >> 1;              // 0..3
  const int wn   = wave & 1;               // 0..1
  const int mbase = blockIdx.x * 128;
  const int nbase = blockIdx.y * 64;

  v8f zero = {};
  v8f acc[2][2];
  acc[0][0] = zero; acc[0][1] = zero; acc[1][0] = zero; acc[1][1] = zero;

  for (int k0 = 0; k0 < K; k0 += 64) {
    __syncthreads();
    // stage A tile (128x64) -> bf16 pairs; 4096 pairs / 256 threads = 16 each
#pragma unroll
    for (int i = 0; i < 16; i++) {
      int p   = tid * 16 + i;
      int row = p >> 5;
      int cp  = p & 31;
      const float2 s = *(const float2*)(A + (mbase + row) * K + k0 + cp * 2);
      Alds[row][cp] = pack2bf(s.x, s.y);
    }
    // stage W^T tile (64n x 64k); 2048 pairs / 256 threads = 8 each
#pragma unroll
    for (int i = 0; i < 8; i++) {
      int p  = tid * 8 + i;
      int n  = p >> 5;
      int cp = p & 31;
      int k  = k0 + cp * 2;
      float w0 = W[k * N + nbase + n];
      float w1 = W[(k + 1) * N + nbase + n];
      Bt[n][cp] = pack2bf(w0, w1);
    }
    __syncthreads();

#pragma unroll
    for (int kf = 0; kf < 2; kf++) {
#pragma unroll
      for (int mt = 0; mt < 2; mt++) {
        FragU a;
        int row = wm * 32 + mt * 16 + r;
#pragma unroll
        for (int j = 0; j < 8; j++) {
          int k2 = g * 16 + 2 * j;
          a.u[j] = Alds[row][kf * 16 + (k2 >> 1)];
        }
#pragma unroll
        for (int nt = 0; nt < 2; nt++) {
          FragU b;
          int n = wn * 32 + nt * 16 + r;
#pragma unroll
          for (int j = 0; j < 8; j++)
            b.u[j] = Bt[n][kf * 16 + g * 8 + j];
          acc[mt][nt] = __builtin_amdgcn_wmma_f32_16x16x32_bf16(
              false, a.v, false, b.v, (short)0, acc[mt][nt], false, false);
        }
      }
    }
  }

#pragma unroll
  for (int mt = 0; mt < 2; mt++) {
#pragma unroll
    for (int j = 0; j < 8; j++) {
      int row = mbase + wm * 32 + mt * 16 + g * 8 + j;
      float sc = cscale * (float)rmask[(row / MDIV) % MMOD];  // constant div/mod
#pragma unroll
      for (int nt = 0; nt < 2; nt++) {
        int col = nbase + wn * 32 + nt * 16 + r;
        C[row * N + col] = acc[mt][nt][j] * sc;
      }
    }
  }
}

// =====================================================================
// Level-1 flash attention per (b,h,l). Softmax over LK per level.
// grid.x = B*H*NL, grid.y = LQ/128; 8 waves, each owns 16 q-rows.
// pq: (B,LQ,D)  pk,pv: (NL,B,LK,D)  mask: (B,LQ,LK) uint8
// O1: (B,LQ,NL,D)
// =====================================================================
__global__ __launch_bounds__(256)
void attn_level1(const float* __restrict__ pq, const float* __restrict__ pk,
                 const float* __restrict__ pv, const unsigned char* __restrict__ maskp,
                 float* __restrict__ O1)
{
  __shared__ uint32_t Klds[32][33];              // [kcol][dpair] : K block, 32x64 bf16
  __shared__ uint32_t Vt[64][17];                // [dcol][kpair] : V^T block, 64x32 bf16
  __shared__ __align__(4) uint16_t Pl[8][16][34];// per-wave P tile, 16x32 bf16

  const int tid  = threadIdx.x;
  const int wave = tid >> 5;
  const int lane = tid & 31;
  const int g    = lane >> 4;
  const int r    = lane & 15;

  const int bx = blockIdx.x;
  const int l  = bx % NLc;
  const int h  = (bx / NLc) % Hc;          // power-of-2 -> cheap
  const int b  = bx / (NLc * Hc);
  const int q0 = blockIdx.y * 128 + wave * 16;

  // Q fragments (16 x 64 = two K=32 A-frags), loaded once per wave
  FragU qa[2];
#pragma unroll
  for (int kf = 0; kf < 2; kf++) {
#pragma unroll
    for (int j = 0; j < 8; j++) {
      int k2 = g * 16 + 2 * j;
      int d  = kf * 32 + k2;
      const float2 s = *(const float2*)(pq + (b * LQc + q0 + r) * Dc + h * DHc + d);
      qa[kf].u[j] = pack2bf(s.x, s.y);
    }
  }

  v8f zero = {};
  v8f acc[4];
  acc[0] = zero; acc[1] = zero; acc[2] = zero; acc[3] = zero;
  float mrow[8], lrow[8];
#pragma unroll
  for (int j = 0; j < 8; j++) { mrow[j] = -3.0e38f; lrow[j] = 0.0f; }

  const int kvbase = (l * Bc + b) * LKc;

  for (int kb = 0; kb < LKc; kb += 32) {
    __syncthreads();
    // stage K block (32 kcols x 64 d) — d pairs contiguous
#pragma unroll
    for (int i = 0; i < 4; i++) {
      int p  = tid * 4 + i;
      int kr = p >> 5;
      int pp = p & 31;
      const float2 s = *(const float2*)(pk + (kvbase + kb + kr) * Dc + h * DHc + pp * 2);
      Klds[kr][pp] = pack2bf(s.x, s.y);
    }
    // stage V^T block (64 d x 32 k) — k pairs contiguous
#pragma unroll
    for (int i = 0; i < 4; i++) {
      int p  = tid * 4 + i;
      int dc = p >> 4;
      int pr = p & 15;
      float v0 = pv[(kvbase + kb + pr * 2    ) * Dc + h * DHc + dc];
      float v1 = pv[(kvbase + kb + pr * 2 + 1) * Dc + h * DHc + dc];
      Vt[dc][pr] = pack2bf(v0, v1);
    }
    __syncthreads();

    // S (16x32) = Q @ K^T : two 16x16 C tiles, contracting d=64 in two K=32 frags
    v8f S[2];
#pragma unroll
    for (int t = 0; t < 2; t++) {
      S[t] = zero;
#pragma unroll
      for (int kf = 0; kf < 2; kf++) {
        FragU bfr;
#pragma unroll
        for (int j = 0; j < 8; j++)
          bfr.u[j] = Klds[t * 16 + r][kf * 16 + g * 8 + j];
        S[t] = __builtin_amdgcn_wmma_f32_16x16x32_bf16(
            false, qa[kf].v, false, bfr.v, (short)0, S[t], false, false);
      }
    }

    // mask (True -> excluded)
#pragma unroll
    for (int t = 0; t < 2; t++)
#pragma unroll
      for (int j = 0; j < 8; j++) {
        int q = q0 + g * 8 + j;
        int k = kb + t * 16 + r;
        if (maskp[(b * LQc + q) * LKc + k]) S[t][j] = -1.0e30f;
      }

    // online softmax: row m = g*8+j lives in 16 lanes of one half-wave
#pragma unroll
    for (int j = 0; j < 8; j++) {
      float rm = fmaxf(S[0][j], S[1][j]);
      rm = fmaxf(rm, __shfl_xor(rm, 1, 16));
      rm = fmaxf(rm, __shfl_xor(rm, 2, 16));
      rm = fmaxf(rm, __shfl_xor(rm, 4, 16));
      rm = fmaxf(rm, __shfl_xor(rm, 8, 16));
      float nm = fmaxf(mrow[j], rm);
      float sc = __expf(mrow[j] - nm);
      mrow[j] = nm;
      float p0 = __expf(S[0][j] - nm);
      float p1 = __expf(S[1][j] - nm);
      S[0][j] = p0; S[1][j] = p1;
      float rs = p0 + p1;
      rs += __shfl_xor(rs, 1, 16);
      rs += __shfl_xor(rs, 2, 16);
      rs += __shfl_xor(rs, 4, 16);
      rs += __shfl_xor(rs, 8, 16);
      lrow[j] = lrow[j] * sc + rs;
#pragma unroll
      for (int t2 = 0; t2 < 4; t2++) acc[t2][j] *= sc;
    }

    // C-layout -> A-fragment layout via per-wave LDS bounce (bf16)
#pragma unroll
    for (int t = 0; t < 2; t++)
#pragma unroll
      for (int j = 0; j < 8; j++)
        Pl[wave][g * 8 + j][t * 16 + r] = (uint16_t)f2bf(S[t][j]);

    FragU pa;
#pragma unroll
    for (int j = 0; j < 8; j++) {
      int k2 = g * 16 + 2 * j;
      pa.u[j] = *(const uint32_t*)&Pl[wave][r][k2];
    }

    // acc += P @ V : four 16x16 output d-tiles, K=32
#pragma unroll
    for (int t2 = 0; t2 < 4; t2++) {
      FragU vb;
#pragma unroll
      for (int j = 0; j < 8; j++)
        vb.u[j] = Vt[t2 * 16 + r][g * 8 + j];
      acc[t2] = __builtin_amdgcn_wmma_f32_16x16x32_bf16(
          false, pa.v, false, vb.v, (short)0, acc[t2], false, false);
    }
  }

  // normalize and store into O1 (B,LQ,NL,D)
#pragma unroll
  for (int j = 0; j < 8; j++) {
    float inv = 1.0f / lrow[j];
    int q = q0 + g * 8 + j;
    int base = ((b * LQc + q) * NLc + l) * Dc + h * DHc;
#pragma unroll
    for (int t2 = 0; t2 < 4; t2++)
      O1[base + t2 * 16 + r] = acc[t2][j] * inv;
  }
}

// =====================================================================
// Level-2 attention: one wave per (b,q,h); softmax over NL=6.
// NQ: (B,LQ,D)  OC: (B,LQ,NL,D)  O2: (B,LQ,D)
// =====================================================================
__global__ __launch_bounds__(256)
void attn_level2(const float* __restrict__ NQ, const float* __restrict__ OC,
                 float* __restrict__ O2)
{
  int idx  = blockIdx.x * 8 + (threadIdx.x >> 5);   // (b*LQ+q)*H + h
  int lane = threadIdx.x & 31;
  int h  = idx & (Hc - 1);
  int bq = idx >> 4;                                 // H == 16

  const float* qp = NQ + bq * Dc + h * DHc;
  float q0 = qp[lane * 2];
  float q1 = qp[lane * 2 + 1];

  float s[NLc], k0c[NLc], k1c[NLc];
#pragma unroll
  for (int lvl = 0; lvl < NLc; lvl++) {
    const float* kp = OC + (bq * NLc + lvl) * Dc + h * DHc;
    float a = kp[lane * 2], c = kp[lane * 2 + 1];
    k0c[lvl] = a; k1c[lvl] = c;
    float part = q0 * a + q1 * c;
#pragma unroll
    for (int m = 16; m >= 1; m >>= 1) part += __shfl_xor(part, m, 32);
    s[lvl] = part;
  }
  float mx = s[0];
#pragma unroll
  for (int lvl = 1; lvl < NLc; lvl++) mx = fmaxf(mx, s[lvl]);
  float sum = 0.f, p[NLc];
#pragma unroll
  for (int lvl = 0; lvl < NLc; lvl++) { p[lvl] = __expf(s[lvl] - mx); sum += p[lvl]; }
  float inv = 1.0f / sum;
  float o0 = 0.f, o1 = 0.f;
#pragma unroll
  for (int lvl = 0; lvl < NLc; lvl++) {
    float w = p[lvl] * inv;
    o0 += w * k0c[lvl];
    o1 += w * k1c[lvl];
  }
  O2[bq * Dc + h * DHc + lane * 2]     = o0;
  O2[bq * Dc + h * DHc + lane * 2 + 1] = o1;
}

// =====================================================================
extern "C" void kernel_launch(void* const* d_in, const int* in_sizes, int n_in,
                              void* d_out, int out_size, void* d_ws, size_t ws_size,
                              hipStream_t stream)
{
  (void)in_sizes; (void)n_in; (void)out_size; (void)ws_size;

  const float*         query  = (const float*)d_in[0];
  const float*         key    = (const float*)d_in[1];
  const unsigned char* mask   = (const unsigned char*)d_in[2];
  const int*           qmask  = (const int*)d_in[3];
  const int*           vmask  = (const int*)d_in[4];
  const float*         W_q    = (const float*)d_in[5];
  const float*         W_k    = (const float*)d_in[6];
  const float*         W_v    = (const float*)d_in[7];
  const float*         W_q2   = (const float*)d_in[8];
  const float*         W_cat  = (const float*)d_in[9];
  const float*         W_cat2 = (const float*)d_in[10];
  float* out = (float*)d_out;

  const size_t nq = (size_t)Bc * LQc * Dc;        // 1.57M floats
  const size_t nk = (size_t)NLc * Bc * LKc * Dc;  // 9.44M floats
  float* pq  = (float*)d_ws;
  float* pk  = pq + nq;
  float* pv  = pk + nk;
  float* O1  = pv + nk;
  float* OC  = pk;   // reuse after attn_level1
  float* NQv = pv;   // reuse after attn_level1
  float* O2  = pq;   // reuse after attn_level1

  dim3 blk(256);

  // projections (row scale fuses qm/vm and dh^-0.5)
  gemm_bf16<1, Bc*LQc><<<dim3((Bc*LQc)/128, Dc/64), blk, 0, stream>>>(
      query, W_q, pq, Dc, Dc, 0.125f, qmask);
  gemm_bf16<1, Bc*LKc><<<dim3((NLc*Bc*LKc)/128, Dc/64), blk, 0, stream>>>(
      key, W_k, pk, Dc, Dc, 1.0f, vmask);
  gemm_bf16<1, Bc*LKc><<<dim3((NLc*Bc*LKc)/128, Dc/64), blk, 0, stream>>>(
      key, W_v, pv, Dc, Dc, 1.0f, vmask);

  // level-1 attention (flash, per (b,h,l))
  attn_level1<<<dim3(Bc*Hc*NLc, LQc/128), blk, 0, stream>>>(pq, pk, pv, mask, O1);

  // out @ W_cat * qm  (row = (b*LQ+q)*NL + l -> qm idx = row/NL)
  gemm_bf16<NLc, Bc*LQc><<<dim3((Bc*LQc*NLc)/128, Dc/64), blk, 0, stream>>>(
      O1, W_cat, OC, Dc, Dc, 1.0f, qmask);

  // nq = query @ W_q2 * qm
  gemm_bf16<1, Bc*LQc><<<dim3((Bc*LQc)/128, Dc/64), blk, 0, stream>>>(
      query, W_q2, NQv, Dc, Dc, 1.0f, qmask);

  // level-2 attention (softmax over NL)
  attn_level2<<<dim3((Bc*LQc*Hc)/8), blk, 0, stream>>>(NQv, OC, O2);

  // out2 @ W_cat2 * qm -> final output
  gemm_bf16<1, Bc*LQc><<<dim3((Bc*LQc)/128, Dc/64), blk, 0, stream>>>(
      O2, W_cat2, out, Dc, Dc, 1.0f, qmask);
}
